// PointNet2SAModuleMSG_5007931867445
// MI455X (gfx1250) — matrix operand order repaired and checked by reading, not compiled, not run on hardware
//
#include <hip/hip_runtime.h>
#include <hip/hip_bf16.h>
#include <cstdint>

// ---------------- problem constants ----------------
#define BB      4
#define NN      8192
#define NPT     2048
#define CIN     64
#define K0PAD   96     // 67 padded up to a multiple of 32

typedef __bf16 v16bf __attribute__((ext_vector_type(16)));
typedef float  v8f   __attribute__((ext_vector_type(8)));

union FragBf { v16bf v; uint4 u[2]; };

__device__ __forceinline__ unsigned short f2bf(float f) {
  unsigned u = __float_as_uint(f);
  unsigned r = u + 0x7FFFu + ((u >> 16) & 1u);   // round-to-nearest-even
  return (unsigned short)(r >> 16);
}
__device__ __forceinline__ float bf2f(unsigned short s) {
  return __uint_as_float(((unsigned)s) << 16);
}

// =====================================================================
// 1) Farthest-point sampling: one 1024-thread workgroup per batch.
//    Points live in registers (8 per thread, layout p = k*1024 + t).
//    Selected point broadcast through LDS by its owner thread.
// =====================================================================
__global__ __launch_bounds__(1024) void fps_kernel(const float* __restrict__ xyz,
                                                   float* __restrict__ new_xyz) {
  const int b = blockIdx.x;
  const int t = threadIdx.x;
  const float* base = xyz + (long)b * NN * 3;

  float px[8], py[8], pz[8], dist[8];
#pragma unroll
  for (int k = 0; k < 8; ++k) {
    int p = k * 1024 + t;
    px[k] = base[p * 3 + 0];
    py[k] = base[p * 3 + 1];
    pz[k] = base[p * 3 + 2];
    dist[k] = 1e10f;
  }

  __shared__ float sp[3];
  __shared__ float rd[32];
  __shared__ int   ri[32];
  __shared__ int   sLast;

  int last = 0;
  for (int it = 0; it < NPT; ++it) {
    if ((last & 1023) == t) {              // owner publishes the point
      int k = last >> 10;
      sp[0] = px[k]; sp[1] = py[k]; sp[2] = pz[k];
      long o = ((long)b * NPT + it) * 3;
      new_xyz[o + 0] = px[k];
      new_xyz[o + 1] = py[k];
      new_xyz[o + 2] = pz[k];
    }
    __syncthreads();
    if (it == NPT - 1) break;

    float qx = sp[0], qy = sp[1], qz = sp[2];
    float bd = -1.0f; int bi = 0x7FFFFFFF;
#pragma unroll
    for (int k = 0; k < 8; ++k) {
      float dx = px[k] - qx, dy = py[k] - qy, dz = pz[k] - qz;
      float d = dx * dx + dy * dy + dz * dz;
      dist[k] = fminf(dist[k], d);
      int p = k * 1024 + t;
      if (dist[k] > bd || (dist[k] == bd && p < bi)) { bd = dist[k]; bi = p; }
    }
    // wave32 reduction (prefer lower index on ties, matching jnp.argmax)
#pragma unroll
    for (int off = 16; off > 0; off >>= 1) {
      float od = __shfl_down(bd, off);
      int   oi = __shfl_down(bi, off);
      if (od > bd || (od == bd && oi < bi)) { bd = od; bi = oi; }
    }
    if ((t & 31) == 0) { rd[t >> 5] = bd; ri[t >> 5] = bi; }
    __syncthreads();
    if (t == 0) {
      float best = rd[0]; int bidx = ri[0];
      for (int w = 1; w < 32; ++w)
        if (rd[w] > best || (rd[w] == best && ri[w] < bidx)) { best = rd[w]; bidx = ri[w]; }
      sLast = bidx;
    }
    __syncthreads();
    last = sLast;
  }
}

// =====================================================================
// 2) Ball query: one wave per query point. Ordered compaction of the
//    first `ns` in-range indices (ascending), pad with first / N-1.
// =====================================================================
__global__ __launch_bounds__(256) void ballquery_kernel(const float* __restrict__ xyz,
                                                        const float* __restrict__ new_xyz,
                                                        int* __restrict__ idx_out,
                                                        int ns, float r2) {
  int q    = (int)((blockIdx.x * blockDim.x + threadIdx.x) >> 5);
  int lane = threadIdx.x & 31;
  if (q >= BB * NPT) return;
  int b = q >> 11;
  const float* base = xyz + (long)b * NN * 3;
  float qx = new_xyz[(long)q * 3 + 0];
  float qy = new_xyz[(long)q * 3 + 1];
  float qz = new_xyz[(long)q * 3 + 2];
  int* out = idx_out + (long)q * ns;

  int cnt = 0;
  int first = NN;
  for (int j0 = 0; j0 < NN && cnt < ns; j0 += 32) {
    int p = j0 + lane;
    float dx = base[p * 3 + 0] - qx;
    float dy = base[p * 3 + 1] - qy;
    float dz = base[p * 3 + 2] - qz;
    bool inR = (dx * dx + dy * dy + dz * dz) <= r2;
    unsigned mask = (unsigned)__ballot(inR);
    if (cnt == 0 && mask) first = j0 + __ffs(mask) - 1;
    int pre = __popc(mask & ((1u << lane) - 1u));
    if (inR && (cnt + pre) < ns) out[cnt + pre] = p;
    cnt += __popc(mask);
  }
  if (cnt > ns) cnt = ns;
  int padv = min(first, NN - 1);
  for (int p = cnt + lane; p < ns; p += 32) out[p] = padv;
}

// =====================================================================
// 3) Gather + concat -> bf16 activation rows [G][96]
//    cols 0..2 = xyz[idx]-new_xyz, 3..66 = features[:,idx], 67..95 = 0
// =====================================================================
__global__ __launch_bounds__(256) void gather_concat_kernel(const float* __restrict__ xyz,
                                                            const float* __restrict__ feats,
                                                            const float* __restrict__ new_xyz,
                                                            const int* __restrict__ idx,
                                                            unsigned short* __restrict__ X0,
                                                            int ns) {
  long g = (long)blockIdx.x * 256 + threadIdx.x;
  long G = (long)BB * NPT * ns;
  if (g >= G) return;
  long bs = g / ns;              // b*NPT + s
  int  b  = (int)(bs >> 11);
  int  pt = idx[g];

  const float* p = xyz + ((long)b * NN + pt) * 3;
  const float* q = new_xyz + bs * 3;

  union { unsigned short row[96]; uint4 v4[12]; } u;
  u.row[0] = f2bf(p[0] - q[0]);
  u.row[1] = f2bf(p[1] - q[1]);
  u.row[2] = f2bf(p[2] - q[2]);
  const float* f = feats + (long)b * CIN * NN + pt;
#pragma unroll
  for (int c = 0; c < CIN; ++c) u.row[3 + c] = f2bf(f[(long)c * NN]);
#pragma unroll
  for (int k = 67; k < 96; ++k) u.row[k] = 0;

  uint4* dst = (uint4*)(X0 + g * 96);   // 192 B per row -> 16 B aligned
#pragma unroll
  for (int i = 0; i < 12; ++i) dst[i] = u.v4[i];
}

// =====================================================================
// 4) WMMA GEMM + BN statistics.
//    Y[G,COUT] = X[G,KPAD] * W^T, W staged in LDS as bf16 [COUT][KPAD].
//    Block = 256 threads (8 waves), each wave owns a 16-row M tile and
//    all COUT columns (COUT/16 accumulator tiles of v_wmma 16x16x32).
// =====================================================================
template <int COUT, int KPAD>
__global__ __launch_bounds__(256) void gemm_bn_kernel(const unsigned short* __restrict__ X,
                                                      const unsigned short* __restrict__ Wb,
                                                      unsigned short* __restrict__ Y,
                                                      float* __restrict__ stats) {
  constexpr int NT = COUT / 16;
  constexpr int KS = KPAD / 32;
  __shared__ unsigned short sW[COUT * KPAD];
  __shared__ float sSum[COUT];
  __shared__ float sSq[COUT];

  int tid = threadIdx.x;
  for (int i = tid; i < COUT * KPAD; i += 256) sW[i] = Wb[i];
  if (tid < COUT) { sSum[tid] = 0.0f; sSq[tid] = 0.0f; }
  __syncthreads();

  int wave = tid >> 5;
  int lane = tid & 31;
  int h    = lane >> 4;        // K-half selector (ISA 16-bit A/B layout)
  int l16  = lane & 15;

  long rowBase = (long)blockIdx.x * 128 + wave * 16;
  const unsigned short* xrow = X + (rowBase + l16) * KPAD;

  v8f acc[NT];
#pragma unroll
  for (int t = 0; t < NT; ++t)
#pragma unroll
    for (int v = 0; v < 8; ++v) acc[t][v] = 0.0f;

#pragma unroll
  for (int ks = 0; ks < KS; ++ks) {
    int kk = ks * 32;
    FragA_load:;
    FragBf a;
    a.u[0] = *(const uint4*)(xrow + kk + 8 * h);        // K = kk+8h .. +7
    a.u[1] = *(const uint4*)(xrow + kk + 16 + 8 * h);   // K = kk+16+8h .. +7
#pragma unroll
    for (int t = 0; t < NT; ++t) {
      FragBf fb;                                         // B[k][n] = W[n][k]
      const unsigned short* wrow = sW + (t * 16 + l16) * KPAD + kk + 8 * h;
      fb.u[0] = *(const uint4*)(wrow);
      fb.u[1] = *(const uint4*)(wrow + 16);
      acc[t] = __builtin_amdgcn_wmma_f32_16x16x32_bf16(
          false, a.v, false, fb.v, (short)0, acc[t], false, false);
    }
  }

  // write bf16 output + per-channel partial sums (LDS atomics, then global)
#pragma unroll
  for (int t = 0; t < NT; ++t) {
    int c = t * 16 + l16;
    float s = 0.0f, sq = 0.0f;
#pragma unroll
    for (int v = 0; v < 8; ++v) {
      float val = acc[t][v];
      long  r   = rowBase + h * 8 + v;
      Y[r * COUT + c] = f2bf(val);
      s  += val;
      sq += val * val;
    }
    atomicAdd(&sSum[c], s);
    atomicAdd(&sSq[c], sq);
  }
  __syncthreads();
  if (tid < COUT) {
    atomicAdd(&stats[tid], sSum[tid]);
    atomicAdd(&stats[COUT + tid], sSq[tid]);
  }
}

// =====================================================================
// 5) BN helpers
// =====================================================================
__global__ void zero_stats_kernel(float* stats) { stats[threadIdx.x] = 0.0f; }

__global__ void bn_finalize_kernel(const float* __restrict__ stats,
                                   const float* __restrict__ gamma,
                                   const float* __restrict__ beta,
                                   float* __restrict__ ab, int C, float invG) {
  int c = threadIdx.x;
  if (c < C) {
    float mu  = stats[c] * invG;
    float var = stats[C + c] * invG - mu * mu;
    float inv = rsqrtf(var + 1e-5f);
    float a = inv * gamma[c];
    ab[c]       = a;
    ab[128 + c] = beta[c] - mu * a;
  }
}

__global__ __launch_bounds__(256) void bn_apply_kernel(unsigned short* __restrict__ Y,
                                                       const float* __restrict__ ab,
                                                       long total, int cmask) {
  long i = (long)blockIdx.x * 256 + threadIdx.x;
  if (i >= total) return;
  int c = (int)(i & cmask);
  float v = bf2f(Y[i]);
  v = fmaxf(v * ab[c] + ab[128 + c], 0.0f);
  Y[i] = f2bf(v);
}

// =====================================================================
// 6) Final-layer BN+ReLU fused with max-pool over ns, f32 writeout.
//    out layout: d_out[24576 + ((b*256 + coutBase + c)*NPT + s)]
// =====================================================================
__global__ __launch_bounds__(256) void pool_out_kernel(const unsigned short* __restrict__ Y,
                                                       const float* __restrict__ ab,
                                                       float* __restrict__ out,
                                                       int ns, int coutBase) {
  long i = (long)blockIdx.x * 256 + threadIdx.x;
  long total = (long)BB * NPT * 128;
  if (i >= total) return;
  int  c  = (int)(i & 127);
  long bs = i >> 7;                    // b*NPT + s
  float a = ab[c], bb = ab[128 + c];
  const unsigned short* p = Y + (bs * ns) * 128 + c;
  float m = -1e30f;
  for (int k = 0; k < ns; ++k) {
    float v = bf2f(p[(long)k * 128]);
    v = fmaxf(v * a + bb, 0.0f);
    m = fmaxf(m, v);
  }
  int b = (int)(bs >> 11);
  int s = (int)(bs & (NPT - 1));
  out[(long)BB * NPT * 3 + ((long)(b * 256 + coutBase + c) * NPT + s)] = m;
}

// =====================================================================
// 7) Weight conversion: f32 [Cout][Cin] -> bf16 [Cout][Kpad] (zero pad)
// =====================================================================
__global__ __launch_bounds__(256) void wprep_kernel(const float* __restrict__ W,
                                                    unsigned short* __restrict__ Wb,
                                                    int Cout, int Cin, int Kpad) {
  int i = blockIdx.x * 256 + threadIdx.x;
  if (i >= Cout * Kpad) return;
  int r = i / Kpad;
  int k = i - r * Kpad;
  Wb[i] = (k < Cin) ? f2bf(W[r * Cin + k]) : (unsigned short)0;
}

// =====================================================================
// host-side orchestration
// =====================================================================
extern "C" void kernel_launch(void* const* d_in, const int* in_sizes, int n_in,
                              void* d_out, int out_size, void* d_ws, size_t ws_size,
                              hipStream_t stream) {
  const float* xyz   = (const float*)d_in[0];
  const float* feats = (const float*)d_in[1];
  float* out = (float*)d_out;

  // workspace layout
  char* ws = (char*)d_ws;
  size_t off = 0;
  int* idxBuf = (int*)(ws + off);                 off += (size_t)BB * NPT * 32 * 4;  // 1 MB
  unsigned short* WbSlot[8];
  for (int i = 0; i < 8; ++i) { WbSlot[i] = (unsigned short*)(ws + off); off += 128 * 128 * 2; }
  float* stats = (float*)(ws + off);              off += 256 * 4;
  float* ab    = (float*)(ws + off);              off += 256 * 4;
  off = (off + 255) & ~(size_t)255;
  unsigned short* bufA = (unsigned short*)(ws + off); off += (size_t)262144 * 128 * 2; // 64 MB
  unsigned short* bufB = (unsigned short*)(ws + off); off += (size_t)262144 * 128 * 2; // 64 MB

  // --- FPS: new_xyz -> d_out[0 .. 24575] ---
  fps_kernel<<<BB, 1024, 0, stream>>>(xyz, out);
  const float* new_xyz = out;

  // --- weight prep (8 matrices) ---
  struct { int din; int cout; int cin; int kpad; } wspec[8] = {
    { 2, 64,  67, K0PAD}, { 5, 64, 64, 64}, { 8, 64, 64, 64}, {11, 128,  64, 64},
    {14, 64,  67, K0PAD}, {17, 64, 64, 64}, {20, 128, 64, 64}, {23, 128, 128, 128},
  };
  for (int i = 0; i < 8; ++i) {
    int n = wspec[i].cout * wspec[i].kpad;
    wprep_kernel<<<(n + 255) / 256, 256, 0, stream>>>(
        (const float*)d_in[wspec[i].din], WbSlot[i],
        wspec[i].cout, wspec[i].cin, wspec[i].kpad);
  }

  // ================= branch 0: r=0.1, ns=16 =================
  {
    const int ns = 16;
    const long G = (long)BB * NPT * ns;          // 131072
    const int gblk = (int)(G / 128);             // 1024
    ballquery_kernel<<<(BB * NPT * 32) / 256, 256, 0, stream>>>(xyz, new_xyz, idxBuf, ns, 0.01f);
    gather_concat_kernel<<<(int)(G / 256), 256, 0, stream>>>(xyz, feats, new_xyz, idxBuf, bufA, ns);
    const float invG = 1.0f / (float)G;

    // L0: 96 -> 64
    zero_stats_kernel<<<1, 256, 0, stream>>>(stats);
    gemm_bn_kernel<64, 96><<<gblk, 256, 0, stream>>>(bufA, WbSlot[0], bufB, stats);
    bn_finalize_kernel<<<1, 128, 0, stream>>>(stats, (const float*)d_in[3], (const float*)d_in[4], ab, 64, invG);
    bn_apply_kernel<<<(int)(G * 64 / 256), 256, 0, stream>>>(bufB, ab, G * 64, 63);
    // L1: 64 -> 64
    zero_stats_kernel<<<1, 256, 0, stream>>>(stats);
    gemm_bn_kernel<64, 64><<<gblk, 256, 0, stream>>>(bufB, WbSlot[1], bufA, stats);
    bn_finalize_kernel<<<1, 128, 0, stream>>>(stats, (const float*)d_in[6], (const float*)d_in[7], ab, 64, invG);
    bn_apply_kernel<<<(int)(G * 64 / 256), 256, 0, stream>>>(bufA, ab, G * 64, 63);
    // L2: 64 -> 64
    zero_stats_kernel<<<1, 256, 0, stream>>>(stats);
    gemm_bn_kernel<64, 64><<<gblk, 256, 0, stream>>>(bufA, WbSlot[2], bufB, stats);
    bn_finalize_kernel<<<1, 128, 0, stream>>>(stats, (const float*)d_in[9], (const float*)d_in[10], ab, 64, invG);
    bn_apply_kernel<<<(int)(G * 64 / 256), 256, 0, stream>>>(bufB, ab, G * 64, 63);
    // L3: 64 -> 128 (BN folded into pool)
    zero_stats_kernel<<<1, 256, 0, stream>>>(stats);
    gemm_bn_kernel<128, 64><<<gblk, 256, 0, stream>>>(bufB, WbSlot[3], bufA, stats);
    bn_finalize_kernel<<<1, 128, 0, stream>>>(stats, (const float*)d_in[12], (const float*)d_in[13], ab, 128, invG);
    pool_out_kernel<<<(BB * NPT * 128) / 256, 256, 0, stream>>>(bufA, ab, out, ns, 0);
  }

  // ================= branch 1: r=0.2, ns=32 =================
  {
    const int ns = 32;
    const long G = (long)BB * NPT * ns;          // 262144
    const int gblk = (int)(G / 128);             // 2048
    ballquery_kernel<<<(BB * NPT * 32) / 256, 256, 0, stream>>>(xyz, new_xyz, idxBuf, ns, 0.04f);
    gather_concat_kernel<<<(int)(G / 256), 256, 0, stream>>>(xyz, feats, new_xyz, idxBuf, bufA, ns);
    const float invG = 1.0f / (float)G;

    // L0: 96 -> 64
    zero_stats_kernel<<<1, 256, 0, stream>>>(stats);
    gemm_bn_kernel<64, 96><<<gblk, 256, 0, stream>>>(bufA, WbSlot[4], bufB, stats);
    bn_finalize_kernel<<<1, 128, 0, stream>>>(stats, (const float*)d_in[15], (const float*)d_in[16], ab, 64, invG);
    bn_apply_kernel<<<(int)(G * 64 / 256), 256, 0, stream>>>(bufB, ab, G * 64, 63);
    // L1: 64 -> 64
    zero_stats_kernel<<<1, 256, 0, stream>>>(stats);
    gemm_bn_kernel<64, 64><<<gblk, 256, 0, stream>>>(bufB, WbSlot[5], bufA, stats);
    bn_finalize_kernel<<<1, 128, 0, stream>>>(stats, (const float*)d_in[18], (const float*)d_in[19], ab, 64, invG);
    bn_apply_kernel<<<(int)(G * 64 / 256), 256, 0, stream>>>(bufA, ab, G * 64, 63);
    // L2: 64 -> 128
    zero_stats_kernel<<<1, 256, 0, stream>>>(stats);
    gemm_bn_kernel<128, 64><<<gblk, 256, 0, stream>>>(bufA, WbSlot[6], bufB, stats);
    bn_finalize_kernel<<<1, 128, 0, stream>>>(stats, (const float*)d_in[21], (const float*)d_in[22], ab, 128, invG);
    bn_apply_kernel<<<(int)(G * 128 / 256), 256, 0, stream>>>(bufB, ab, G * 128, 127);
    // L3: 128 -> 128 (BN folded into pool)
    zero_stats_kernel<<<1, 256, 0, stream>>>(stats);
    gemm_bn_kernel<128, 128><<<gblk, 256, 0, stream>>>(bufB, WbSlot[7], bufA, stats);
    bn_finalize_kernel<<<1, 128, 0, stream>>>(stats, (const float*)d_in[24], (const float*)d_in[25], ab, 128, invG);
    pool_out_kernel<<<(BB * NPT * 128) / 256, 256, 0, stream>>>(bufA, ab, out, ns, 128);
  }
}
